// Constellation_CNN_26139170964092
// MI455X (gfx1250) — matrix operand order repaired and checked by reading, not compiled, not run on hardware
//
#include <hip/hip_runtime.h>
#include <hip/hip_fp16.h>

// ---------------------------------------------------------------------------
// Types for WMMA fragments (gfx1250, wave32)
// ---------------------------------------------------------------------------
typedef __attribute__((ext_vector_type(16))) _Float16 v16h;
typedef __attribute__((ext_vector_type(8)))  float    v8f;

union ABFrag { v16h v; uint4 q[2]; };

__device__ __forceinline__ v16h load_frag(const _Float16* p0, const _Float16* p1) {
    ABFrag t;
    t.q[0] = *(const uint4*)p0;
    t.q[1] = *(const uint4*)p1;
    return t.v;
}
__device__ __forceinline__ float prelu(float x, float a) { return x > 0.f ? x : a * x; }

// ---------------------------------------------------------------------------
// Problem constants
// ---------------------------------------------------------------------------
#define BATCH 1024
#define SLEN  4096

// Workspace layout (element offsets of the f16 zero page relative to h1 / h2)
#define WS_IMG    0
#define WS_H1     16777216
#define WS_H2     83886080
#define WS_H3F    117440512
#define WS_FC1    121634816
#define WS_WB2    122159104
#define WS_WB3    122195968
#define WS_WC1F   122605568
#define WS_WC2F   123654144
#define WS_ZPAD   123719680
#define ZOFF_H1   ((WS_ZPAD - WS_H1) / 2)   // 53,471,232 f16 elements
#define ZOFF_H2   ((WS_ZPAD - WS_H2) / 2)   // 19,916,800 f16 elements

// ---------------------------------------------------------------------------
// 0) zero helper (images + zero-page)
// ---------------------------------------------------------------------------
__global__ void zero_f4_kernel(float4* __restrict__ p, int n4) {
    int t = blockIdx.x * 256 + threadIdx.x;
    if (t < n4) p[t] = make_float4(0.f, 0.f, 0.f, 0.f);
}

// ---------------------------------------------------------------------------
// 1) IQ -> constellation occupancy grid. All colliding writes store 1.0f.
// ---------------------------------------------------------------------------
__global__ void scatter_kernel(const float* __restrict__ x, float* __restrict__ img) {
    int t = blockIdx.x * 256 + threadIdx.x;      // B*S threads
    int s = t & (SLEN - 1);
    int b = t >> 12;
    float xi = x[((size_t)b * 2 + 0) * SLEN + s];
    float xq = x[((size_t)b * 2 + 1) * SLEN + s];
    int ii = (int)(xi * 16.f + 32.f); ii = ii < 0 ? 0 : (ii > 63 ? 63 : ii);
    int qi = (int)(xq * 16.f + 32.f); qi = qi < 0 ? 0 : (qi > 63 ? 63 : qi);
    img[((size_t)b * 64 + qi) * 64 + ii] = 1.0f;
}

// ---------------------------------------------------------------------------
// 2) weight transforms: OIHW fp32 -> [tap][och][ch] f16 (GEMM-B layout)
// ---------------------------------------------------------------------------
__global__ void cvt_w2_kernel(const float* __restrict__ w2, _Float16* __restrict__ dst) {
    int d = blockIdx.x * 256 + threadIdx.x;          // 9*64*32
    if (d >= 9 * 64 * 32) return;
    int tap = d / 2048, r = d % 2048, o = r >> 5, c = r & 31;
    int dy = tap / 3, dx = tap % 3;
    dst[d] = (_Float16)w2[((o * 32 + c) * 3 + dy) * 3 + dx];
}
__global__ void cvt_w3_kernel(const float* __restrict__ w3, _Float16* __restrict__ dst) {
    int d = blockIdx.x * 256 + threadIdx.x;          // 25*128*64
    if (d >= 25 * 128 * 64) return;
    int tap = d / 8192, r = d % 8192, o = r >> 6, c = r & 63;
    int dy = tap / 5, dx = tap % 5;
    dst[d] = (_Float16)w3[((o * 64 + c) * 5 + dy) * 5 + dx];
}
__global__ void cvt_flat_kernel(const float* __restrict__ src, _Float16* __restrict__ dst, int n) {
    int t = blockIdx.x * 256 + threadIdx.x;
    if (t < n) dst[t] = (_Float16)src[t];
}

// ---------------------------------------------------------------------------
// 3) conv1 (1->32ch, 3x3 pad1) + bias + PReLU + avgpool2, fused, scalar.
//    Output h1: NHWC f16 [B][32][32][32]
// ---------------------------------------------------------------------------
__global__ __launch_bounds__(256)
void conv1_fused_kernel(const float* __restrict__ img, const float* __restrict__ w1,
                        const float* __restrict__ b1, const float* __restrict__ a1p,
                        _Float16* __restrict__ h1) {
    int t  = blockIdx.x * 256 + threadIdx.x;     // B*32*32*32
    int oc = t & 31;
    int xp = (t >> 5) & 31;
    int yp = (t >> 10) & 31;
    int b  = t >> 15;
    const float* ip = img + (size_t)b * 4096;
    int y0 = yp * 2 - 1, x0 = xp * 2 - 1;
    float v[4][4];
#pragma unroll
    for (int i = 0; i < 4; ++i)
#pragma unroll
        for (int j = 0; j < 4; ++j) {
            int y = y0 + i, x = x0 + j;
            v[i][j] = ((unsigned)y < 64u && (unsigned)x < 64u) ? ip[y * 64 + x] : 0.f;
        }
    float w[9];
#pragma unroll
    for (int k = 0; k < 9; ++k) w[k] = w1[oc * 9 + k];
    float bias = b1[oc], al = a1p[0];
    float acc = 0.f;
#pragma unroll
    for (int py = 0; py < 2; ++py)
#pragma unroll
        for (int px = 0; px < 2; ++px) {
            float s = bias;
#pragma unroll
            for (int ky = 0; ky < 3; ++ky)
#pragma unroll
                for (int kx = 0; kx < 3; ++kx) s += v[py + ky][px + kx] * w[ky * 3 + kx];
            acc += prelu(s, al);
        }
    h1[((size_t)(b * 32 + yp) * 32 + xp) * 32 + oc] = (_Float16)(acc * 0.25f);
}

// ---------------------------------------------------------------------------
// 4) conv2 (32->64ch, 3x3 pad1) as implicit GEMM via WMMA f16, fused
//    bias + PReLU + avgpool2. One wave per (image, 2x8 pixel block, 16-och tile).
//    Out-of-bounds taps resolve to a zero page via a 32-bit offset select
//    (no exec-mask branching, no predicated address blocks).
//    Output h2: NHWC f16 [B][16][16][64]
// ---------------------------------------------------------------------------
__global__ __launch_bounds__(256)
void conv2_wmma_kernel(const _Float16* __restrict__ h1, const _Float16* __restrict__ wB2,
                       const float* __restrict__ b2, const float* __restrict__ a2p,
                       _Float16* __restrict__ h2) {
    const int lane  = threadIdx.x & 31;
    const int wid   = blockIdx.x * 8 + (threadIdx.x >> 5);
    const int ntile = wid & 3;          // 4 och tiles
    const int mtile = (wid >> 2) & 63;  // 64 pixel tiles
    const int b     = wid >> 8;
    const int by = mtile >> 2, bx = mtile & 3;
    const int row = lane & 15, hi = lane >> 4, col = lane & 15;
    const int yy = by * 2 + (row >> 3);
    const int xx = bx * 8 + (row & 7);
    // per-lane base element offset of this pixel's channel vector (incl. K half)
    const int pixBase = ((b * 32 + yy) * 32 + xx) * 32 + hi * 8;
    const int zoff    = ZOFF_H1 + hi * 8;
    const _Float16* pb0 = wB2 + ((size_t)ntile * 16 + col) * 32 + hi * 8;

    v8f c = {};
#pragma unroll
    for (int tap = 0; tap < 9; ++tap) {
        const int dy = tap / 3 - 1, dx = tap % 3 - 1;
        const int iy = yy + dy, ix = xx + dx;
        const bool inb = (unsigned)iy < 32u && (unsigned)ix < 32u;
        const int off = inb ? (pixBase + (dy * 32 + dx) * 32) : zoff;   // v_cndmask
        const _Float16* pa = h1 + off;
        v16h a  = load_frag(pa, pa + 16);
        v16h bf = load_frag(pb0 + tap * 2048, pb0 + tap * 2048 + 16);
        c = __builtin_amdgcn_wmma_f32_16x16x32_f16(false, a, false, bf, (short)0, c, false, false);
    }
    const int och = ntile * 16 + col;
    const float bias = b2[och], al = a2p[0];
    float s[4];
#pragma unroll
    for (int j = 0; j < 4; ++j)
        s[j] = prelu(c[2 * j] + bias, al) + prelu(c[2 * j + 1] + bias, al);
#pragma unroll
    for (int j = 0; j < 4; ++j) s[j] += __shfl_xor(s[j], 16, 32);  // y-pair across lane halves
    if (lane < 16) {
        _Float16* out = h2 + ((size_t)(b * 16 + by) * 16 + bx * 4) * 64 + och;
#pragma unroll
        for (int j = 0; j < 4; ++j) out[j * 64] = (_Float16)(s[j] * 0.25f);
    }
}

// ---------------------------------------------------------------------------
// 5) conv3 (64->128ch, 5x5 pad2) implicit GEMM + bias + PReLU + double avgpool2
//    fused as one mean over each 4x4 pixel block (one 16-row M tile).
//    Output h3f: [B][2048] f16 in CHW-flatten order k = och*16 + by*4 + bx.
// ---------------------------------------------------------------------------
__global__ __launch_bounds__(256)
void conv3_wmma_kernel(const _Float16* __restrict__ h2, const _Float16* __restrict__ wB3,
                       const float* __restrict__ b3, const float* __restrict__ a3p,
                       _Float16* __restrict__ h3f) {
    const int lane  = threadIdx.x & 31;
    const int wid   = blockIdx.x * 8 + (threadIdx.x >> 5);
    const int ntile = wid & 7;          // 8 och tiles (128 ch)
    const int mtile = (wid >> 3) & 15;  // 16 pixel tiles (4x4 blocks)
    const int b     = wid >> 7;
    const int by = mtile >> 2, bx = mtile & 3;
    const int row = lane & 15, hi = lane >> 4, col = lane & 15;
    const int yy = by * 4 + (row >> 2);
    const int xx = bx * 4 + (row & 3);
    const int pixBase = ((b * 16 + yy) * 16 + xx) * 64 + hi * 8;
    const int zoff    = ZOFF_H2 + hi * 8;
    const _Float16* pb0 = wB3 + ((size_t)ntile * 16 + col) * 64 + hi * 8;

    v8f c = {};
#pragma unroll
    for (int tap = 0; tap < 25; ++tap) {
        const int dy = tap / 5 - 2, dx = tap % 5 - 2;
        const int iy = yy + dy, ix = xx + dx;
        const bool inb = (unsigned)iy < 16u && (unsigned)ix < 16u;
        const int off = inb ? (pixBase + (dy * 16 + dx) * 64) : zoff;   // v_cndmask
        const _Float16* pa = h2 + off;
        const _Float16* pb = pb0 + tap * 8192;
#pragma unroll
        for (int kk = 0; kk < 2; ++kk) {
            v16h a  = load_frag(pa + kk * 32,      pa + kk * 32 + 16);
            v16h bf = load_frag(pb + kk * 32,      pb + kk * 32 + 16);
            c = __builtin_amdgcn_wmma_f32_16x16x32_f16(false, a, false, bf, (short)0, c, false, false);
        }
    }
    const int och = ntile * 16 + col;
    const float bias = b3[och], al = a3p[0];
    float sum = 0.f;
#pragma unroll
    for (int v = 0; v < 8; ++v) sum += prelu(c[v] + bias, al);
    sum += __shfl_xor(sum, 16, 32);
    if (lane < 16)
        h3f[(size_t)b * 2048 + och * 16 + by * 4 + bx] = (_Float16)(sum * 0.0625f);
}

// ---------------------------------------------------------------------------
// 6) FC1: [1024,2048] x [2048,256]^T + bias + PReLU -> f16 [1024,256]
// ---------------------------------------------------------------------------
__global__ __launch_bounds__(256)
void fc1_wmma_kernel(const _Float16* __restrict__ h3f, const _Float16* __restrict__ wc1f,
                     const float* __restrict__ bc1, const float* __restrict__ acp,
                     _Float16* __restrict__ fc1out) {
    const int lane  = threadIdx.x & 31;
    const int wid   = blockIdx.x * 8 + (threadIdx.x >> 5);
    const int ntile = wid & 15;   // 16 tiles of 16 och
    const int mtile = wid >> 4;   // 64 tiles of 16 batch rows
    const int row = lane & 15, hi = lane >> 4, col = lane & 15;
    const _Float16* pa0 = h3f  + (size_t)(mtile * 16 + row) * 2048 + hi * 8;
    const _Float16* pb0 = wc1f + (size_t)(ntile * 16 + col) * 2048 + hi * 8;
    v8f c = {};
#pragma unroll 4
    for (int kc = 0; kc < 64; ++kc) {
        v16h a  = load_frag(pa0 + kc * 32, pa0 + kc * 32 + 16);
        v16h bf = load_frag(pb0 + kc * 32, pb0 + kc * 32 + 16);
        c = __builtin_amdgcn_wmma_f32_16x16x32_f16(false, a, false, bf, (short)0, c, false, false);
    }
    const int och = ntile * 16 + col;
    const float bias = bc1[och], al = acp[0];
#pragma unroll
    for (int v = 0; v < 8; ++v) {
        int bb = mtile * 16 + v + hi * 8;
        fc1out[(size_t)bb * 256 + och] = (_Float16)prelu(c[v] + bias, al);
    }
}

// ---------------------------------------------------------------------------
// 7) FC2: [1024,256] x [256,128]^T + bias -> fp32 output
// ---------------------------------------------------------------------------
__global__ __launch_bounds__(256)
void fc2_wmma_kernel(const _Float16* __restrict__ fc1out, const _Float16* __restrict__ wc2f,
                     const float* __restrict__ bc2, float* __restrict__ out) {
    const int lane  = threadIdx.x & 31;
    const int wid   = blockIdx.x * 8 + (threadIdx.x >> 5);
    const int ntile = wid & 7;    // 8 tiles of 16 och
    const int mtile = wid >> 3;   // 64 tiles of 16 batch rows
    const int row = lane & 15, hi = lane >> 4, col = lane & 15;
    const _Float16* pa0 = fc1out + (size_t)(mtile * 16 + row) * 256 + hi * 8;
    const _Float16* pb0 = wc2f   + (size_t)(ntile * 16 + col) * 256 + hi * 8;
    v8f c = {};
#pragma unroll
    for (int kc = 0; kc < 8; ++kc) {
        v16h a  = load_frag(pa0 + kc * 32, pa0 + kc * 32 + 16);
        v16h bf = load_frag(pb0 + kc * 32, pb0 + kc * 32 + 16);
        c = __builtin_amdgcn_wmma_f32_16x16x32_f16(false, a, false, bf, (short)0, c, false, false);
    }
    const int och = ntile * 16 + col;
    const float bias = bc2[och];
#pragma unroll
    for (int v = 0; v < 8; ++v) {
        int bb = mtile * 16 + v + hi * 8;
        out[(size_t)bb * 128 + och] = c[v] + bias;
    }
}

// ---------------------------------------------------------------------------
// Host launcher
// ---------------------------------------------------------------------------
extern "C" void kernel_launch(void* const* d_in, const int* in_sizes, int n_in,
                              void* d_out, int out_size, void* d_ws, size_t ws_size,
                              hipStream_t stream) {
    const float* x   = (const float*)d_in[0];
    const float* w1  = (const float*)d_in[1];
    const float* b1  = (const float*)d_in[2];
    const float* a1  = (const float*)d_in[3];
    const float* w2  = (const float*)d_in[4];
    const float* b2  = (const float*)d_in[5];
    const float* a2  = (const float*)d_in[6];
    const float* w3  = (const float*)d_in[7];
    const float* b3  = (const float*)d_in[8];
    const float* a3  = (const float*)d_in[9];
    const float* wc1 = (const float*)d_in[10];
    const float* bc1 = (const float*)d_in[11];
    const float* ac  = (const float*)d_in[12];
    const float* wc2 = (const float*)d_in[13];
    const float* bc2 = (const float*)d_in[14];
    float* out = (float*)d_out;

    // Workspace layout (all offsets 256-B aligned)
    char* w = (char*)d_ws;
    float*    img    = (float*)   (w + WS_IMG);   // 1024*64*64 f32      = 16 MiB
    _Float16* h1     = (_Float16*)(w + WS_H1);    // 1024*32*32*32 f16   = 64 MiB
    _Float16* h2     = (_Float16*)(w + WS_H2);    // 1024*16*16*64 f16   = 32 MiB
    _Float16* h3f    = (_Float16*)(w + WS_H3F);   // 1024*2048 f16       =  4 MiB
    _Float16* fc1out = (_Float16*)(w + WS_FC1);   // 1024*256 f16        = 512 KiB
    _Float16* wB2    = (_Float16*)(w + WS_WB2);   // 9*64*32 f16
    _Float16* wB3    = (_Float16*)(w + WS_WB3);   // 25*128*64 f16
    _Float16* wc1f   = (_Float16*)(w + WS_WC1F);  // 256*2048 f16
    _Float16* wc2f   = (_Float16*)(w + WS_WC2F);  // 128*256 f16
    _Float16* zpad   = (_Float16*)(w + WS_ZPAD);  // 256 B zero page

    // 0) zero images + zero page
    zero_f4_kernel<<<4096, 256, 0, stream>>>((float4*)img, 1024 * 64 * 64 / 4);
    zero_f4_kernel<<<1, 256, 0, stream>>>((float4*)zpad, 16);
    // weight transforms (independent of img)
    cvt_w2_kernel<<<72, 256, 0, stream>>>(w2, wB2);
    cvt_w3_kernel<<<800, 256, 0, stream>>>(w3, wB3);
    cvt_flat_kernel<<<2048, 256, 0, stream>>>(wc1, wc1f, 256 * 2048);
    cvt_flat_kernel<<<128, 256, 0, stream>>>(wc2, wc2f, 128 * 256);
    // 1) scatter
    scatter_kernel<<<16384, 256, 0, stream>>>(x, img);
    // 3) conv1 fused (B*32*32*32 threads)
    conv1_fused_kernel<<<131072, 256, 0, stream>>>(img, w1, b1, a1, h1);
    // 4) conv2 WMMA: 1024 imgs * 64 mtiles * 4 ntiles waves / 8 per block
    conv2_wmma_kernel<<<32768, 256, 0, stream>>>(h1, wB2, b2, a2, h2);
    // 5) conv3 WMMA: 1024 * 16 * 8 waves / 8
    conv3_wmma_kernel<<<16384, 256, 0, stream>>>(h2, wB3, b3, a3, h3f);
    // 6) FC1: 64*16 waves / 8
    fc1_wmma_kernel<<<128, 256, 0, stream>>>(h3f, wc1f, bc1, ac, fc1out);
    // 7) FC2: 64*8 waves / 8
    fc2_wmma_kernel<<<64, 256, 0, stream>>>(fc1out, wc2f, bc2, out);

    (void)in_sizes; (void)n_in; (void)out_size; (void)ws_size;
}
